// FDT_26834955665624
// MI455X (gfx1250) — matrix operand order repaired and checked by reading, not compiled, not run on hardware
//
#include <hip/hip_runtime.h>
#include <hip/hip_bf16.h>

typedef __bf16 bf16;
typedef __attribute__((ext_vector_type(16))) __bf16 bf16x16;
typedef __attribute__((ext_vector_type(8)))  float   f32x8;

#define PTOT   65536      // B*H*W
#define CCH    128
#define HIDP   384        // HID=340 padded to multiple of 64 (uniform strides)

// ---------------- workspace layout (bytes, all 256-aligned) ----------------
#define OFF_A   ((size_t)0)              // f32 [P][128]  residual stream x
#define OFF_B   ((size_t)33554432)       // f32 [P][128]  x_ (branch output)
#define OFF_T   ((size_t)67108864)       // bf16 [P][128] LN out / attn out
#define OFF_Q   ((size_t)83886080)       // bf16 [P][384] qkv lin  (reused: FFN hidden)
#define OFF_R   ((size_t)134217728)      // bf16 [P][384] qkv dw   (reused: FFN hidden2)
#define OFF_W   ((size_t)184549376)      // bf16 weights
#define OFF_PN  ((size_t)185204736)      // f32 [2][16][256] norm partials
#define OFF_PA  ((size_t)185237504)      // f32 [2][4][64][1024] attn partials
#define OFF_AF  ((size_t)187334656)      // f32 [2][4][1024] final attn

// weight sub-offsets (bytes from OFF_W)
#define W_RSA_QKV 0
#define W_RSA_PRJ 98304
#define W_FFS_IN  131072
#define W_FFS_OUT 229376
#define W_GSA_QKV 327680
#define W_GSA_PRJ 425984
#define W_FFC_IN  458752
#define W_FFC_OUT 557056

// ---------------------------------------------------------------------------
// weight fp32 -> bf16 with zero padding to [rowsP][colsP]
__global__ void cvtw_k(const float* __restrict__ in, bf16* __restrict__ out,
                       int rows, int cols, int rowsP, int colsP) {
  int idx = blockIdx.x * 256 + threadIdx.x;
  if (idx >= rowsP * colsP) return;
  int r = idx / colsP, c = idx - r * colsP;
  out[idx] = (r < rows && c < cols) ? (bf16)in[r * cols + c] : (bf16)0.0f;
}

// NCHW f32 -> NHWC f32
__global__ void tr_k(const float* __restrict__ X, float* __restrict__ A) {
  int idx = blockIdx.x * 256 + threadIdx.x;          // 8,388,608 total
  int b = idx >> 21, c = (idx >> 14) & 127, q = idx & 16383;
  A[(size_t)((b << 14) + q) * CCH + c] = X[idx];
}

// per-pixel channel LayerNorm (wave per pixel); optional residual add A += Badd
__global__ __launch_bounds__(256) void ln_k(float* __restrict__ A,
                                            const float* __restrict__ Badd,
                                            const float* __restrict__ w,
                                            const float* __restrict__ bia,
                                            bf16* __restrict__ T, int addB) {
  int lane = threadIdx.x & 31, wave = threadIdx.x >> 5;
  int p = blockIdx.x * 8 + wave;
  size_t base = (size_t)p * CCH + lane * 4;
  float4 xv = *(const float4*)(A + base);
  if (addB) {
    float4 bv = *(const float4*)(Badd + base);
    xv.x += bv.x; xv.y += bv.y; xv.z += bv.z; xv.w += bv.w;
    *(float4*)(A + base) = xv;
  }
  float s  = xv.x + xv.y + xv.z + xv.w;
  float s2 = xv.x*xv.x + xv.y*xv.y + xv.z*xv.z + xv.w*xv.w;
  #pragma unroll
  for (int off = 16; off >= 1; off >>= 1) { s += __shfl_xor(s, off); s2 += __shfl_xor(s2, off); }
  float m = s * (1.f/128.f);
  float rs = rsqrtf(s2 * (1.f/128.f) - m*m + 1e-6f);
  int c = lane * 4;
  union { bf16 h[4]; uint2 u; } o;
  o.h[0] = (bf16)((xv.x - m) * rs * w[c+0] + bia[c+0]);
  o.h[1] = (bf16)((xv.y - m) * rs * w[c+1] + bia[c+1]);
  o.h[2] = (bf16)((xv.z - m) * rs * w[c+2] + bia[c+2]);
  o.h[3] = (bf16)((xv.w - m) * rs * w[c+3] + bia[c+3]);
  *(uint2*)(T + base) = o.u;
}

// ---------------------------------------------------------------------------
// bf16 WMMA GEMM: OUT[p][n] = sum_k X[p][k] * W[n][k]    (NHWC x [O][Cin])
// wave tile: 32 pixels x 32 out-channels (2x2 fragments, 4 accumulators).
// CIN is a template param -> K-loop fully unrolled, no register rotation movs;
// the scheduler hoists next-step load clauses over the WMMA quads (partial
// s_wait_loadcnt), overlapping VMEM with the matrix pipe.
// EP: 0=store bf16   1=store f32   2=f32 + res   3=NCHW f32 + res (final out)
union U16 { bf16x16 v; uint4 u[2]; };

__device__ __forceinline__ U16 ldA(const bf16* base, int kb) {
  U16 r;
  r.u[0] = *(const uint4*)(base + kb);
  r.u[1] = *(const uint4*)(base + kb + 16);
  return r;
}
__device__ __forceinline__ U16 ldB(const bf16* base, int kb) {
  U16 r;
  r.u[0] = *(const uint4*)(base + kb);
  r.u[1] = *(const uint4*)(base + kb + 16);
  return r;
}

template<int CIN, int EP>
__global__ __launch_bounds__(256) void gemm_bf16_wmma(
    const bf16* __restrict__ X, const bf16* __restrict__ W, int Cout,
    bf16* __restrict__ outb, float* __restrict__ outf,
    const float* __restrict__ res) {
  int lane = threadIdx.x & 31, wave = threadIdx.x >> 5;
  int l15 = lane & 15;
  int m0 = (blockIdx.x * 8 + wave) * 32;
  int n0 = blockIdx.y * 32;
  // A fragments: lane<16 -> K {0..7,16..23}; lane>=16 -> K {8..15,24..31}
  const bf16* Ar0 = X + (size_t)(m0 + l15) * CIN + ((lane < 16) ? 0 : 8);
  const bf16* Ar1 = Ar0 + (size_t)16 * CIN;
  // B fragments: lane<16 -> K {0..7,16..23} of col n; lane>=16 -> K {8..15,24..31}
  const bf16* Bc0 = W + (size_t)(n0 + l15) * CIN + ((lane < 16) ? 0 : 8);
  const bf16* Bc1 = Bc0 + (size_t)16 * CIN;
  if (EP >= 2) __builtin_prefetch(res + (size_t)m0 * CCH + n0);  // global_prefetch_b8

  f32x8 acc00 = {0.f,0.f,0.f,0.f,0.f,0.f,0.f,0.f};
  f32x8 acc01 = acc00, acc10 = acc00, acc11 = acc00;

  #pragma unroll
  for (int kb = 0; kb < CIN; kb += 32) {
    U16 a0 = ldA(Ar0, kb), a1 = ldA(Ar1, kb);
    U16 b0 = ldB(Bc0, kb), b1 = ldB(Bc1, kb);
    acc00 = __builtin_amdgcn_wmma_f32_16x16x32_bf16(false, a0.v, false, b0.v, (short)0, acc00, false, false);
    acc01 = __builtin_amdgcn_wmma_f32_16x16x32_bf16(false, a0.v, false, b1.v, (short)0, acc01, false, false);
    acc10 = __builtin_amdgcn_wmma_f32_16x16x32_bf16(false, a1.v, false, b0.v, (short)0, acc10, false, false);
    acc11 = __builtin_amdgcn_wmma_f32_16x16x32_bf16(false, a1.v, false, b1.v, (short)0, acc11, false, false);
  }

  #pragma unroll
  for (int mi = 0; mi < 2; mi++) {
    int mb = m0 + mi * 16 + ((lane < 16) ? 0 : 8);
    #pragma unroll
    for (int f = 0; f < 2; f++) {
      int n = n0 + f * 16 + l15;
      f32x8 acc = mi ? (f ? acc11 : acc10) : (f ? acc01 : acc00);
      #pragma unroll
      for (int r = 0; r < 8; r++) {
        int p = mb + r;
        float v = acc[r];
        if (EP == 0)      outb[(size_t)p * Cout + n] = (bf16)v;
        else if (EP == 1) outf[(size_t)p * Cout + n] = v;
        else if (EP == 2) outf[(size_t)p * Cout + n] = v + res[(size_t)p * CCH + n];
        else {            // NCHW final store with residual
          int b = p >> 14;
          outf[((size_t)(b * CCH + n) << 14) + (p & 16383)] = v + res[(size_t)p * CCH + n];
        }
      }
    }
  }
}

// ---------------------------------------------------------------------------
// depthwise 3x3, SAME zero padding, optional circular roll (applied in gather)
// and optional exact GELU. 8 channels per thread, NHWC bf16, stride 384.
__global__ __launch_bounds__(256) void dwconv_k(const bf16* __restrict__ I,
                                                bf16* __restrict__ O,
                                                const float* __restrict__ Wd,
                                                int Cw, int rollS, int doGelu) {
  int t = blockIdx.x * 256 + threadIdx.x;
  int p = t / 48;                 // 384/8 = 48 channel-groups per pixel
  int c0 = (t - p * 48) << 3;
  int b = p >> 14, yy = (p >> 7) & 127, xx = p & 127;
  float acc[8] = {0,0,0,0,0,0,0,0};
  for (int dy = 0; dy < 3; dy++) {
    int ny = yy + dy - 1;
    if ((unsigned)ny >= 128u) continue;
    for (int dx = 0; dx < 3; dx++) {
      int nx = xx + dx - 1;
      if ((unsigned)nx >= 128u) continue;
      int sy = (ny + rollS) & 127, sx = (nx + rollS) & 127;
      size_t sp = ((size_t)(b << 14) + (sy << 7) + sx) * 384 + c0;
      union { uint4 u; bf16 h[8]; } v;
      v.u = *(const uint4*)(I + sp);
      int tap = dy * 3 + dx;
      #pragma unroll
      for (int j = 0; j < 8; j++) {
        int c = c0 + j;
        if (c < Cw) acc[j] += (float)v.h[j] * Wd[c * 9 + tap];
      }
    }
  }
  union { uint4 u; bf16 h[8]; } o;
  #pragma unroll
  for (int j = 0; j < 8; j++) {
    float v = acc[j];
    if (doGelu) v = 0.5f * v * (1.f + erff(v * 0.70710678118f));
    o.h[j] = (c0 + j < Cw) ? (bf16)v : (bf16)0.0f;
  }
  *(uint4*)(O + (size_t)p * 384 + c0) = o.u;
}

// ---------------------------------------------------------------------------
// RSA window attention: one workgroup per 4x4 window (rolled space).
// q,k l2norm over C per pixel; attn[c][d]=relu(t*sum_p q k); out=v@attn.
// Output written with inverse roll (+2,+2).
__global__ __launch_bounds__(256) void rsa_attn_k(const bf16* __restrict__ R,
                                                  const float* __restrict__ temp,
                                                  bf16* __restrict__ T) {
  __shared__ bf16 sqkv[16][384];
  __shared__ bf16 sattn[128][128];
  __shared__ float sinv[32];
  int wid = blockIdx.x;
  int b = wid >> 10, w2 = wid & 1023, wy = w2 >> 5, wx = w2 & 31;
  for (int j = threadIdx.x; j < 16 * 384; j += 256) {
    int pi = j / 384, c = j - pi * 384;
    int ry = wy * 4 + (pi >> 2), rx = wx * 4 + (pi & 3);
    sqkv[pi][c] = R[(size_t)((b << 14) + (ry << 7) + rx) * 384 + c];
  }
  __syncthreads();
  if (threadIdx.x < 32) {
    int sel = threadIdx.x >> 4, pi = threadIdx.x & 15;
    float s = 0;
    for (int c = 0; c < 128; c++) { float f = (float)sqkv[pi][sel * 128 + c]; s += f * f; }
    sinv[threadIdx.x] = 1.f / fmaxf(sqrtf(s), 1e-12f);
  }
  __syncthreads();
  for (int j = threadIdx.x; j < 16 * 256; j += 256) {
    int pi = j >> 8, c = j & 255, sel = c >> 7;
    sqkv[pi][c] = (bf16)((float)sqkv[pi][c] * sinv[sel * 16 + pi]);
  }
  __syncthreads();
  float tmp = temp[0];
  {
    int c = threadIdx.x >> 1, d0 = (threadIdx.x & 1) * 64;
    float qn[16];
    #pragma unroll
    for (int pi = 0; pi < 16; pi++) qn[pi] = (float)sqkv[pi][c];
    for (int d = d0; d < d0 + 64; d++) {
      float s = 0;
      #pragma unroll
      for (int pi = 0; pi < 16; pi++) s += qn[pi] * (float)sqkv[pi][128 + d];
      sattn[c][d] = (bf16)fmaxf(s * tmp, 0.f);
    }
  }
  __syncthreads();
  {
    int pi = threadIdx.x >> 4, d0 = (threadIdx.x & 15) * 8;
    float acc[8] = {0,0,0,0,0,0,0,0};
    for (int c = 0; c < 128; c++) {
      float vv = (float)sqkv[pi][256 + c];
      #pragma unroll
      for (int j = 0; j < 8; j++) acc[j] += vv * (float)sattn[c][d0 + j];
    }
    int ry = wy * 4 + (pi >> 2), rx = wx * 4 + (pi & 3);
    int py = (ry + 2) & 127, px = (rx + 2) & 127;   // inverse roll
    size_t base = (size_t)((b << 14) + (py << 7) + px) * CCH + d0;
    #pragma unroll
    for (int j = 0; j < 8; j++) T[base + j] = (bf16)acc[j];
  }
}

// ---------------------------------------------------------------------------
// GSA: deterministic partial reductions (no float atomics)
__global__ __launch_bounds__(256) void gsa_norms_part_k(const bf16* __restrict__ R,
                                                        float* __restrict__ pn) {
  int b = blockIdx.x >> 4, slab = blockIdx.x & 15, c = threadIdx.x;  // c: 0..255 (q:0-127,k:128-255)
  float s = 0;
  size_t base = ((size_t)(b << 14) + slab * 1024) * 384 + c;
  for (int i = 0; i < 1024; i++) { float f = (float)R[base + (size_t)i * 384]; s += f * f; }
  pn[(b * 16 + slab) * 256 + c] = s;
}

__global__ __launch_bounds__(256) void gsa_attn_part_k(const bf16* __restrict__ R,
                                                       float* __restrict__ pa) {
  int id = blockIdx.x;                       // 2 x 4 x 64
  int b = id >> 8, hz = id & 255, h = hz >> 6, slab = hz & 63;
  __shared__ bf16 sq[256][32], sk[256][32];
  int p0 = (b << 14) + slab * 256;
  for (int j = threadIdx.x; j < 256 * 32; j += 256) {
    int pi = j >> 5, c = j & 31;
    size_t rb = (size_t)(p0 + pi) * 384 + h * 32 + c;
    sq[pi][c] = R[rb];
    sk[pi][c] = R[rb + 128];
  }
  __syncthreads();
  int c = threadIdx.x >> 3, d0 = (threadIdx.x & 7) * 4;
  float s0 = 0, s1 = 0, s2 = 0, s3 = 0;
  for (int pi = 0; pi < 256; pi++) {
    float qv = (float)sq[pi][c];
    s0 += qv * (float)sk[pi][d0 + 0];
    s1 += qv * (float)sk[pi][d0 + 1];
    s2 += qv * (float)sk[pi][d0 + 2];
    s3 += qv * (float)sk[pi][d0 + 3];
  }
  float* dst = pa + (((size_t)(b * 4 + h) * 64 + slab) * 1024 + c * 32 + d0);
  dst[0] = s0; dst[1] = s1; dst[2] = s2; dst[3] = s3;
}

__global__ __launch_bounds__(256) void gsa_finalize_k(const float* __restrict__ pa,
                                                      const float* __restrict__ pn,
                                                      const float* __restrict__ temp,
                                                      float* __restrict__ af) {
  int b = blockIdx.x >> 2, h = blockIdx.x & 3;
  __shared__ float siq[32], sik[32];
  if (threadIdx.x < 64) {
    int sel = threadIdx.x >> 5, i = threadIdx.x & 31;
    int c = sel * 128 + h * 32 + i;           // q or k channel
    float s = 0;
    for (int sl = 0; sl < 16; sl++) s += pn[(b * 16 + sl) * 256 + c];
    float inv = 1.f / fmaxf(sqrtf(s), 1e-12f);
    if (sel == 0) siq[i] = inv; else sik[i] = inv;
  }
  __syncthreads();
  float tmp = temp[0];
  for (int e = threadIdx.x * 4; e < (threadIdx.x + 1) * 4; e++) {
    int cl = e >> 5, d = e & 31;
    float s = 0;
    for (int sl = 0; sl < 64; sl++) s += pa[((size_t)(b * 4 + h) * 64 + sl) * 1024 + e];
    af[(size_t)(b * 4 + h) * 1024 + e] = fmaxf(s * siq[cl] * sik[d] * tmp, 0.f);
  }
}

__global__ __launch_bounds__(256) void gsa_apply_k(const bf16* __restrict__ R,
                                                   const float* __restrict__ af,
                                                   bf16* __restrict__ T) {
  int p0 = blockIdx.x * 64;
  int b = p0 >> 14;
  __shared__ float sA[4096];
  __shared__ bf16 sV[64][128];
  for (int j = threadIdx.x; j < 4096; j += 256) sA[j] = af[(size_t)b * 4096 + j];
  for (int j = threadIdx.x; j < 64 * 128; j += 256) {
    int pi = j >> 7, c = j & 127;
    sV[pi][c] = R[(size_t)(p0 + pi) * 384 + 256 + c];
  }
  __syncthreads();
  for (int j = threadIdx.x; j < 8192; j += 256) {
    int pi = j >> 7, c = j & 127, h = c >> 5, cl = c & 31;
    const float* arow = sA + h * 1024 + cl * 32;
    float s = 0;
    #pragma unroll
    for (int d = 0; d < 32; d++) s += arow[d] * (float)sV[pi][h * 32 + d];
    T[(size_t)(p0 + pi) * CCH + c] = (bf16)s;
  }
}

// ---------------------------------------------------------------------------
extern "C" void kernel_launch(void* const* d_in, const int* in_sizes, int n_in,
                              void* d_out, int out_size, void* d_ws, size_t ws_size,
                              hipStream_t stream) {
  (void)in_sizes; (void)n_in; (void)out_size; (void)ws_size;
  const float* x        = (const float*)d_in[0];
  const float* w_s0     = (const float*)d_in[1];
  const float* b_s0     = (const float*)d_in[2];
  const float* w_s2     = (const float*)d_in[3];
  const float* b_s2     = (const float*)d_in[4];
  const float* rsa_qkv  = (const float*)d_in[5];
  const float* rsa_dw   = (const float*)d_in[6];
  const float* rsa_proj = (const float*)d_in[7];
  const float* rsa_temp = (const float*)d_in[8];
  const float* ffs_in   = (const float*)d_in[9];
  const float* ffs_dw   = (const float*)d_in[10];
  const float* ffs_out  = (const float*)d_in[11];
  const float* w_c0     = (const float*)d_in[12];
  const float* b_c0     = (const float*)d_in[13];
  const float* w_c2     = (const float*)d_in[14];
  const float* b_c2     = (const float*)d_in[15];
  const float* gsa_qkv  = (const float*)d_in[16];
  const float* gsa_dw   = (const float*)d_in[17];
  const float* gsa_proj = (const float*)d_in[18];
  const float* gsa_temp = (const float*)d_in[19];
  const float* ffc_in   = (const float*)d_in[20];
  const float* ffc_dw   = (const float*)d_in[21];
  const float* ffc_out  = (const float*)d_in[22];

  char* ws = (char*)d_ws;
  float* A  = (float*)(ws + OFF_A);
  float* Bb = (float*)(ws + OFF_B);
  bf16*  T  = (bf16*)(ws + OFF_T);
  bf16*  Q  = (bf16*)(ws + OFF_Q);
  bf16*  R  = (bf16*)(ws + OFF_R);
  char*  Wb = ws + OFF_W;
  float* PN = (float*)(ws + OFF_PN);
  float* PA = (float*)(ws + OFF_PA);
  float* AF = (float*)(ws + OFF_AF);
  float* OUT = (float*)d_out;

  auto cvt = [&](const float* in, size_t off, int rows, int cols, int rowsP, int colsP) {
    int n = rowsP * colsP;
    cvtw_k<<<(n + 255) / 256, 256, 0, stream>>>(in, (bf16*)(Wb + off), rows, cols, rowsP, colsP);
  };
  cvt(rsa_qkv,  W_RSA_QKV, 384, 128, 384, 128);
  cvt(rsa_proj, W_RSA_PRJ, 128, 128, 128, 128);
  cvt(ffs_in,   W_FFS_IN,  340, 128, HIDP, 128);
  cvt(ffs_out,  W_FFS_OUT, 128, 340, 128, HIDP);
  cvt(gsa_qkv,  W_GSA_QKV, 384, 128, 384, 128);
  cvt(gsa_proj, W_GSA_PRJ, 128, 128, 128, 128);
  cvt(ffc_in,   W_FFC_IN,  340, 128, HIDP, 128);
  cvt(ffc_out,  W_FFC_OUT, 128, 340, 128, HIDP);

  dim3 gB(PTOT / 32 / 8, 0);
  #define GEMM(CIN, EP, Xp, Woff, Cout, Ob, Of, Rs)                             \
    do { dim3 g(PTOT / 32 / 8, (Cout) / 32);                                    \
         gemm_bf16_wmma<CIN, EP><<<g, 256, 0, stream>>>(                        \
             Xp, (const bf16*)(Wb + (Woff)), Cout, Ob, Of, Rs); } while (0)

  // ---- spatial (RSA) block ----
  tr_k<<<(PTOT * CCH) / 256, 256, 0, stream>>>(x, A);                      // A = x (NHWC)
  ln_k<<<PTOT / 8, 256, 0, stream>>>(A, nullptr, w_s0, b_s0, T, 0);        // T = LN(x)
  GEMM(128, 0, T, W_RSA_QKV, 384, Q, nullptr, nullptr);                    // Q = qkv lin
  dwconv_k<<<(PTOT * 48) / 256, 256, 0, stream>>>(Q, R, rsa_dw, 384, 2, 0);// R = dw (rolled)
  rsa_attn_k<<<2048, 256, 0, stream>>>(R, rsa_temp, T);                    // T = attn out (unrolled)
  GEMM(128, 1, T, W_RSA_PRJ, 128, nullptr, Bb, nullptr);                   // Bb = x_
  ln_k<<<PTOT / 8, 256, 0, stream>>>(A, Bb, w_s2, b_s2, T, 1);             // A = x_+x ; T = LN(A)
  GEMM(128, 0, T, W_FFS_IN, HIDP, Q, nullptr, nullptr);                    // Q = hidden lin
  dwconv_k<<<(PTOT * 48) / 256, 256, 0, stream>>>(Q, R, ffs_dw, 340, 0, 1);// R = gelu(dw)
  GEMM(HIDP, 2, R, W_FFS_OUT, 128, nullptr, A, Bb);                        // A = ffn + x_

  // ---- channel (GSA) block ----
  ln_k<<<PTOT / 8, 256, 0, stream>>>(A, nullptr, w_c0, b_c0, T, 0);        // T = LN(x)
  GEMM(128, 0, T, W_GSA_QKV, 384, Q, nullptr, nullptr);                    // Q = qkv lin
  dwconv_k<<<(PTOT * 48) / 256, 256, 0, stream>>>(Q, R, gsa_dw, 384, 0, 0);// R = dw
  gsa_norms_part_k<<<32, 256, 0, stream>>>(R, PN);
  gsa_attn_part_k<<<512, 256, 0, stream>>>(R, PA);
  gsa_finalize_k<<<8, 256, 0, stream>>>(PA, PN, gsa_temp, AF);
  gsa_apply_k<<<PTOT / 64, 256, 0, stream>>>(R, AF, T);                    // T = attn out
  GEMM(128, 1, T, W_GSA_PRJ, 128, nullptr, Bb, nullptr);                   // Bb = x_
  ln_k<<<PTOT / 8, 256, 0, stream>>>(A, Bb, w_c2, b_c2, T, 1);             // A = x_+x ; T = LN(A)
  GEMM(128, 0, T, W_FFC_IN, HIDP, Q, nullptr, nullptr);                    // Q = hidden lin
  dwconv_k<<<(PTOT * 48) / 256, 256, 0, stream>>>(Q, R, ffc_dw, 340, 0, 1);// R = gelu(dw)
  GEMM(HIDP, 3, R, W_FFC_OUT, 128, nullptr, OUT, Bb);                      // d_out = ffn + x_ (NCHW)
  #undef GEMM
}
